// MultiDimBehavioralCRSAE_16552803958954
// MI455X (gfx1250) — matrix operand order repaired and checked by reading, not compiled, not run on hardware
//
#include <hip/hip_runtime.h>
#include <math.h>
#include <stdint.h>

typedef __attribute__((ext_vector_type(16))) _Float16 v16h;
typedef __attribute__((ext_vector_type(8)))  float    v8f;
typedef unsigned int uint32x4 __attribute__((ext_vector_type(4)));
typedef int          int32x8  __attribute__((ext_vector_type(8)));
typedef int          int32x4  __attribute__((ext_vector_type(4)));

namespace {
constexpr int   YCH  = 8;                 // electrode channels
constexpr int   NTR  = 16;                // trials -> WMMA N columns in synthesis
constexpr int   LLEN = 4096;              // trial length
constexpr int   NF   = 16;                // filters -> WMMA M rows in analysis
constexpr int   KS   = 64;                // kernel taps
constexpr int   ENC  = LLEN - KS + 1;     // 4033
constexpr int   TIT  = 10;                // FISTA iterations
constexpr int   QPAD = 96;                // padded taps/filter (96 = 3*32 chunks align)
constexpr int   NQC  = NF * QPAD / 32;    // 48 synthesis K-chunks
constexpr float INV_LIP = 0.1f;
constexpr float THRESH  = 0.01f;          // lambda / L
}

// ---------------------------------------------------------------------------
// Zero-init x_old, x_tmp (f32) and the f16 x mirror (contiguous in ws).
// ---------------------------------------------------------------------------
__global__ __launch_bounds__(256) void crsae_zero(uint32_t* p, long long nwords) {
  long long i = (long long)blockIdx.x * blockDim.x + threadIdx.x;
  const long long stride = (long long)gridDim.x * blockDim.x;
  for (; i < nwords; i += stride) p[i] = 0u;
}

// ---------------------------------------------------------------------------
// Precompute the synthesis Toeplitz A operand in WMMA *fragment order*:
//   afrag[ch][qc][lane][i] , i-th f16 half of lane's v16h A fragment.
//   A[j, q]=H[f, j+63-u] (zero-padded), q = f*96+u, row j = lane&15,
//   half i -> K = i + 16*(lane>>4).
// ---------------------------------------------------------------------------
__global__ __launch_bounds__(256) void crsae_build_afrag(const float* __restrict__ H,
                                                         _Float16* __restrict__ afrag) {
  const int ch = blockIdx.x;
  for (int e = threadIdx.x; e < NQC * 32 * 16; e += 256) {
    const int qc   = e >> 9;            // 512 halves per chunk
    const int r    = e & 511;
    const int lane = r >> 4;
    const int i    = r & 15;
    const int m    = lane & 15;         // A row (t-offset j)
    const int hi   = lane >> 4;
    const int kin  = i + 16 * hi;
    const int q    = qc * 32 + kin;
    const int f    = q / QPAD;
    const int u    = q - f * QPAD;
    const int tap  = m + 63 - u;        // k = (t0+j) - (t0-63+u)
    float v = 0.0f;
    if (tap >= 0 && tap < KS) v = H[(size_t)(ch * NF + f) * KS + tap];
    afrag[(size_t)ch * (NQC * 512) + e] = (_Float16)v;
  }
}

// ---------------------------------------------------------------------------
// Precompute the analysis A operand (H itself) in fragment order:
//   hfrag[ch][kc][lane][i], row f = lane&15, K = kc*32 + kin.
// ---------------------------------------------------------------------------
__global__ __launch_bounds__(256) void crsae_build_hfrag(const float* __restrict__ H,
                                                         _Float16* __restrict__ hfrag) {
  const int ch = blockIdx.x;
  for (int e = threadIdx.x; e < 2 * 32 * 16; e += 256) {
    const int kc   = e >> 9;
    const int r    = e & 511;
    const int lane = r >> 4;
    const int i    = r & 15;
    const int hi   = lane >> 4;
    const int f    = lane & 15;
    const int kin  = i + 16 * hi;
    hfrag[(size_t)ch * 1024 + e] = (_Float16)H[(size_t)(ch * NF + f) * KS + kc * 32 + kin];
  }
}

// ---------------------------------------------------------------------------
// Synthesis + residual: res16[ch,n,t] = y[n,ch,t] - sum_{f,k} H[f,k]*x[f,t-k]
// GEMM: D[16 t-offsets x 16 trials] = A_toep[16 x 1536] * Xslab[1536 x 16]
// Interior tiles stage the x slab with one Tensor-Data-Mover descriptor
// (tensor_load_to_lds, 3-D tile: 96 t x 16 f x 16 n -> contiguous LDS);
// edge tiles use a scalar zero-padded path. A fragments stream from ws
// (global_load_b128), B fragments are aligned v16h ds_load_b128.
// grid: (LLEN/16, YCH), block: 32 (one wave)
// ---------------------------------------------------------------------------
__global__ __launch_bounds__(32) void crsae_synth(const _Float16* __restrict__ x16,
                                                  const _Float16* __restrict__ afrag,
                                                  const float* __restrict__ y,
                                                  _Float16* __restrict__ res16) {
  __shared__ __align__(128) _Float16 Xs[NTR][NF * QPAD];  // [n][f*96+u], 48 KB

  const int ch   = blockIdx.y;
  const int t0   = blockIdx.x * 16;
  const int lane = threadIdx.x;         // single wave32

  // Prefetch the y rows consumed in the epilogue.
  __builtin_prefetch(&y[((size_t)(lane & 15) * YCH + ch) * LLEN + t0 + (lane >> 4) * 8], 0, 3);

  // Interior tile <=> every (n,f) row span [t0-63, t0+32] lies inside [0, ENC).
  bool tdm_done = false;
#if __has_builtin(__builtin_amdgcn_tensor_load_to_lds) && __has_builtin(__builtin_amdgcn_s_wait_tensorcnt)
  if (t0 >= 64 && t0 + 32 < ENC) {
    const _Float16* gsrc = x16 + (size_t)(ch * NTR) * NF * ENC + (t0 - 63);
    const unsigned long long ga = (unsigned long long)(uintptr_t)gsrc;
    const unsigned lds = (unsigned)(uintptr_t)&Xs[0][0];   // LDS byte offset (addr[31:0])

    uint32x4 g0;
    g0[0] = 1u;                                        // count=1, user descriptor
    g0[1] = lds;                                       // lds_addr
    g0[2] = (unsigned)(ga & 0xffffffffull);            // global_addr[31:0]
    g0[3] = (unsigned)((ga >> 32) & 0x1ffffffull)      // global_addr[56:32]
            | 0x80000000u;                             // type = 2 ("image")

    int32x8 g1;
    g1[0] = 0x00010000;                 // wg_mask=0, data_size=1 (2B), no flags/pad
    g1[1] = (QPAD & 0xffff) << 16;      // tensor_dim0[15:0]  (96)
    g1[2] = (NF << 16);                 // tensor_dim0 hi=0 | tensor_dim1[15:0]=16
    g1[3] = (QPAD << 16);               // tensor_dim1 hi=0 | tile_dim0=96
    g1[4] = NF | (NTR << 16);           // tile_dim1=16 (f) | tile_dim2=16 (n)
    g1[5] = ENC;                        // tensor_dim0_stride[31:0] = 4033
    g1[6] = ((NF * ENC) & 0xffff) << 16;// stride0 hi=0 | tensor_dim1_stride[15:0]
    g1[7] = (NF * ENC) >> 16;           // tensor_dim1_stride[47:16]

    int32x4 g2;
    g2[0] = NTR;                        // tensor_dim2 = 16 (n)
    g2[1] = 0; g2[2] = 0; g2[3] = 0;    // tensor_dim3 / stride2 / tile_dim3 unused
    int32x4 g3 = {0, 0, 0, 0};
    int32x8 g4 = {0, 0, 0, 0, 0, 0, 0, 0};   // extra group (clang-23 form), zero-filled

    __builtin_amdgcn_tensor_load_to_lds(g0, g1, g2, g3, g4, 0);
    __builtin_amdgcn_s_wait_tensorcnt(0);
    tdm_done = true;
  }
#endif
  if (!tdm_done) {
    // Scalar zero-padded staging for boundary tiles.
    for (int n = 0; n < NTR; ++n) {
      const _Float16* xrow = x16 + (size_t)(ch * NTR + n) * NF * ENC;
      for (int f = 0; f < NF; ++f) {
#pragma unroll
        for (int k = 0; k < 3; ++k) {
          const int u = lane + 32 * k;
          const int t = t0 - 63 + u;
          _Float16 v = (_Float16)0.0f;
          if (u < 80 && t >= 0 && t < ENC) v = xrow[f * ENC + t];
          Xs[n][f * QPAD + u] = v;
        }
      }
    }
  }
  __syncthreads();

  const int lo = lane & 15;             // B column: trial ; D column: trial
  const int hi = lane >> 4;
  const _Float16* ap = afrag + (size_t)ch * (NQC * 512) + lane * 16;

  v8f c = {0.f, 0.f, 0.f, 0.f, 0.f, 0.f, 0.f, 0.f};
#pragma unroll 4
  for (int qc = 0; qc < NQC; ++qc) {
    const v16h a = *(const v16h*)(ap + (size_t)qc * 512);          // 2x global_load_b128
    const v16h b = *(const v16h*)&Xs[lo][qc * 32 + 16 * hi];       // 2x ds_load_b128
    c = __builtin_amdgcn_wmma_f32_16x16x32_f16(false, a, false, b,
                                               (short)0, c, false, false);
  }

  // D: column n = lane&15 (trial), row m = r + 8*hi (t-offset).
#pragma unroll
  for (int r = 0; r < 8; ++r) {
    const int t = t0 + r + 8 * hi;
    const float yy = y[((size_t)lo * YCH + ch) * LLEN + t];
    res16[(size_t)(ch * NTR + lo) * LLEN + t] = (_Float16)(yy - c[r]);
  }
}

// ---------------------------------------------------------------------------
// Analysis + fused FISTA update:
//   D[f, j] = sum_k H[f,k] * res[t0+j+k]   (2 WMMAs, K=64)
//   x_new   = relu(x_tmp + D/LIP - THRESH)
//   x_tmp'  = x_new + mom*(x_new - x_old)   (+ f16 mirror for synthesis)
// grid: (LLEN/128, YCH*NTR), block: 256 = 8 waves, one t-tile per wave
// ---------------------------------------------------------------------------
__global__ __launch_bounds__(256) void crsae_analysis(const _Float16* __restrict__ res16,
                                                      const _Float16* __restrict__ hfrag,
                                                      float* xtmp,        // read x_tmp / write x_tmp_next
                                                      const float* xold,  // read x_old (may alias xnew)
                                                      float* xnew,        // buf_old, or d_out on last step
                                                      _Float16* x16,      // f16 mirror of x_tmp_next
                                                      float mom) {
  __shared__ __align__(128) _Float16 Rs[208];   // residual window

  const int cn   = blockIdx.y;                  // ch*16 + n
  const int ch   = cn >> 4;
  const int base = blockIdx.x * 128;
  const int tid  = threadIdx.x;

  for (int i = tid; i < 208; i += 256) {
    const int t = base + i;
    Rs[i] = (t < LLEN) ? res16[(size_t)cn * LLEN + t] : (_Float16)0.0f;
  }
  __syncthreads();

  const int lane = tid & 31;
  const int wave = tid >> 5;                    // tile 0..7
  const int t0   = base + wave * 16;
  const int lo   = lane & 15;
  const int hi   = lane >> 4;
  if (t0 >= ENC) return;                        // wave-uniform: EXEC all-1 for WMMA

  const _Float16* hp = hfrag + (size_t)ch * 1024 + lane * 16;

  v8f c = {0.f, 0.f, 0.f, 0.f, 0.f, 0.f, 0.f, 0.f};
#pragma unroll
  for (int kc = 0; kc < 2; ++kc) {
    const v16h a = *(const v16h*)(hp + kc * 512);       // fragment-order H
    v16h b;
#pragma unroll
    for (int i = 0; i < 16; ++i) {                      // B col j = lo, K = i + 16*hi
      b[i] = Rs[wave * 16 + lo + kc * 32 + i + 16 * hi];
    }
    c = __builtin_amdgcn_wmma_f32_16x16x32_f16(false, a, false, b,
                                               (short)0, c, false, false);
  }

  // D: column j = lane&15 -> t = t0 + j ; rows f = r + 8*hi.
  const int t = t0 + lo;
  if (t < ENC) {
#pragma unroll
    for (int r = 0; r < 8; ++r) {
      const int f = r + 8 * hi;
      const size_t idx = ((size_t)cn * NF + f) * ENC + t;
      const float g  = c[r] * INV_LIP;
      const float xt = xtmp[idx];
      const float xo = xold[idx];
      float xn = xt + g - THRESH;
      xn = xn > 0.0f ? xn : 0.0f;
      const float xtn = xn + mom * (xn - xo);
      xnew[idx] = xn;                    // becomes x_old next step / final output
      xtmp[idx] = xtn;                   // x_tmp_next (same thread, read-before-write)
      x16[idx]  = (_Float16)xtn;         // f16 mirror for the synthesis B operand
    }
  }
}

// ---------------------------------------------------------------------------
extern "C" void kernel_launch(void* const* d_in, const int* in_sizes, int n_in,
                              void* d_out, int out_size, void* d_ws, size_t ws_size,
                              hipStream_t stream) {
  const float* y = (const float*)d_in[0];       // [16, 8, 4096]
  const float* H = (const float*)d_in[1];       // [8, 16, 1, 64]
  float* out = (float*)d_out;                   // [8, 16, 16, 4033]
  char*  ws  = (char*)d_ws;

  const size_t XN   = (size_t)YCH * NTR * NF * ENC;   // 8,259,584
  const size_t RESN = (size_t)YCH * NTR * LLEN;       // 524,288

  size_t off = 0;
  float*    buf_old = (float*)(ws + off);    off += XN * 4;        // x_old / x_new
  float*    buf_tmp = (float*)(ws + off);    off += XN * 4;        // x_tmp / x_tmp_next
  _Float16* x16     = (_Float16*)(ws + off); off += XN * 2;        // f16 mirror of x_tmp
  _Float16* res16   = (_Float16*)(ws + off); off += RESN * 2;      // residual (f16)
  _Float16* afrag   = (_Float16*)(ws + off); off += (size_t)YCH * NQC * 512 * 2;
  _Float16* hfrag   = (_Float16*)(ws + off); off += (size_t)YCH * 1024 * 2;

  // Fragment-order operand tables (once per launch).
  crsae_build_afrag<<<YCH, 256, 0, stream>>>(H, afrag);
  crsae_build_hfrag<<<YCH, 256, 0, stream>>>(H, hfrag);

  // x_old = x_tmp = x16 = 0 (contiguous: 2*XN f32 + XN f16).
  const long long zwords = (long long)(XN * 4 * 2 + XN * 2) / 4;
  crsae_zero<<<2048, 256, 0, stream>>>((uint32_t*)ws, zwords);

  float s_old = 1.0f;
  for (int it = 0; it < TIT; ++it) {
    const float s_new = 0.5f * (1.0f + sqrtf(1.0f + 4.0f * s_old * s_old));
    const float mom   = (s_old - 1.0f) / s_new;          // 0 on first step
    s_old = s_new;

    crsae_synth<<<dim3(LLEN / 16, YCH), 32, 0, stream>>>(x16, afrag, y, res16);

    float* xnew_dst = (it == TIT - 1) ? out : buf_old;
    crsae_analysis<<<dim3(LLEN / 128, YCH * NTR), 256, 0, stream>>>(
        res16, hfrag, buf_tmp, buf_old, xnew_dst, x16, mom);
  }

  (void)in_sizes; (void)n_in; (void)out_size; (void)ws_size;
}
